// RayTransformer_57234734186754
// MI455X (gfx1250) — matrix operand order repaired and checked by reading, not compiled, not run on hardware
//
#include <hip/hip_runtime.h>
#include <hip/hip_bf16.h>
#include <math.h>

typedef __attribute__((ext_vector_type(16))) _Float16 v16h;
typedef __attribute__((ext_vector_type(8)))  float    v8f;

namespace {
constexpr int kNV = 8, kRN = 2048, kSN = 64, kC = 32;
constexpr int kP  = kRN * kSN;        // 131072 points
constexpr int kD  = 72, kL = 65;      // drt token dim / token count
constexpr int kH  = 256, kW = 256, kR = 48;
}

// ---------------- WMMA fragment helpers (gfx1250 wave32 layouts) -------------
__device__ inline v16h frag_a(const _Float16* s, int lane, int rt, int kt, int ld) {
  // A 16x32 f16: lane<16 row=lane holds k 0..15; lane>=16 holds k 16..31 (contiguous per lane)
  int r  = rt * 16 + (lane & 15);
  int kb = kt * 32 + ((lane >> 4) & 1) * 16;
  const _Float16* p = s + r * ld + kb;
  v16h a;
#pragma unroll
  for (int i = 0; i < 16; ++i) a[i] = p[i];
  return a;
}
__device__ inline v16h frag_b(const _Float16* w, int lane, int nt, int kt, int ld) {
  // B 32x16 f16 from weight (Npad x Kpad row-major, B[k][n] = W[n][k]):
  // lane<16: n=lane, k 0..15 ; lane>=16: n=lane-16, k 16..31 (contiguous per lane)
  int n  = nt * 16 + (lane & 15);
  int kb = kt * 32 + ((lane >> 4) & 1) * 16;
  const _Float16* p = w + n * ld + kb;
  v16h b;
#pragma unroll
  for (int i = 0; i < 16; ++i) b[i] = p[i];
  return b;
}
__device__ inline v8f wmma_mk(const _Float16* A, const _Float16* W, int lane,
                              int rt, int nt, int kpad) {
  v8f c = {};
  for (int kt = 0; kt < kpad / 32; ++kt) {
    v16h a = frag_a(A, lane, rt, kt, kpad);
    v16h b = frag_b(W, lane, nt, kt, kpad);
    c = __builtin_amdgcn_wmma_f32_16x16x32_f16(false, a, false, b, (short)0, c,
                                               false, false);
  }
  return c;
}
// Unconditional D-fragment store into a tile-padded buffer: no divergent guards,
// no exec-mask chains. Destination rows/cols beyond the logical matrix are
// padding that consumers never read.
template <bool RELU, bool F16OUT>
__device__ inline void tile_store_full(v8f c, int lane, int rt, int nt,
                                       void* out, int ldo) {
  int col = nt * 16 + (lane & 15);
  int rb  = ((lane >> 4) & 1) * 8;   // VGPR j: row j (lanes<16) / j+8 (lanes>=16)
#pragma unroll
  for (int j = 0; j < 8; ++j) {
    int r = rt * 16 + rb + j;
    float v = c[j];
    if (RELU) v = fmaxf(v, 0.f);
    if (F16OUT) ((_Float16*)out)[r * ldo + col] = (_Float16)v;
    else        ((float*)out)[r * ldo + col] = v;
  }
}
__device__ inline float elup(float x) { return x > 0.f ? x + 1.f : expf(x); }

// ---------------- weight packing: f32 (N,K) -> f16 (Npad,Kpad) zero-padded ---
__global__ __launch_bounds__(256) void pack_w_kernel(const float* __restrict__ w,
                                                     _Float16* __restrict__ o,
                                                     int N, int K, int Np, int Kp) {
  int i = blockIdx.x * blockDim.x + threadIdx.x;
  if (i >= Np * Kp) return;
  int n = i / Kp, k = i - n * Kp;
  o[i] = (_Float16)((n < N && k < K) ? w[n * K + k] : 0.f);
}

// ---------------- geometry + 2D sampling + dir_relative ----------------------
__global__ __launch_bounds__(256) void geom_kernel(
    const float* __restrict__ p3d, const float* __restrict__ simg,
    const float* __restrict__ sfeat, const float* __restrict__ refpi,
    const float* __restrict__ spi, const float* __restrict__ sp,
    float* __restrict__ X0, float* __restrict__ rgbb, float* __restrict__ maskb,
    float* __restrict__ dirb, float* __restrict__ ppix) {
  int t = blockIdx.x * blockDim.x + threadIdx.x;
  if (t >= kP * kNV) return;
  int p = t >> 3, v = t & 7;
  int r = p / kSN, s = p - r * kSN;
  float X = p3d[p * 3 + 0], Y = p3d[p * 3 + 1], Z = p3d[p * 3 + 2];
  const float* M = sp + v * 16;
  float t0 = M[0] * X + M[1] * Y + M[2] * Z + M[3];
  float t1 = M[4] * X + M[5] * Y + M[6] * Z + M[7];
  float t2 = M[8] * X + M[9] * Y + M[10] * Z + M[11];
  float px = t0 / t2, py = t1 / t2;
  ppix[((size_t)(v * 2 + 0) * kRN + r) * kSN + s] = px;
  ppix[((size_t)(v * 2 + 1) * kRN + r) * kSN + s] = py;
  float inb = (px >= 0.f && px <= (float)(kW - 1) && py >= 0.f && py <= (float)(kH - 1)) ? 1.f : 0.f;
  maskb[p * kNV + v] = (t2 > 0.f) ? inb : 0.f;
  float ix = fminf(fmaxf(px, -1.f), (float)kW);
  float iy = fminf(fmaxf(py, -1.f), (float)kH);
  float x0 = floorf(ix), y0 = floorf(iy);
  float wx = ix - x0, wy = iy - y0;
  float cwt[4] = {(1 - wx) * (1 - wy), wx * (1 - wy), (1 - wx) * wy, wx * wy};
  int xi[4], yi[4]; float cv[4];
  for (int cc = 0; cc < 4; ++cc) {
    float xf = x0 + (float)(cc & 1), yf = y0 + (float)(cc >> 1);
    cv[cc] = (xf >= 0.f && xf <= (float)(kW - 1) && yf >= 0.f && yf <= (float)(kH - 1)) ? 1.f : 0.f;
    xi[cc] = (int)fminf(fmaxf(xf, 0.f), (float)(kW - 1));
    yi[cc] = (int)fminf(fmaxf(yf, 0.f), (float)(kH - 1));
  }
  float* xo = X0 + ((size_t)p * 9 + 1 + v) * kC;
  for (int ch = 0; ch < kC; ++ch) {
    const float* bimg = sfeat + (size_t)(v * kC + ch) * kH * kW;
    float acc = 0.f;
    for (int cc = 0; cc < 4; ++cc) acc += cv[cc] * cwt[cc] * bimg[yi[cc] * kW + xi[cc]];
    xo[ch] = acc;
  }
  for (int ch = 0; ch < 3; ++ch) {
    const float* bimg = simg + (size_t)(v * 3 + ch) * kH * kW;
    float acc = 0.f;
    for (int cc = 0; cc < 4; ++cc) acc += cv[cc] * cwt[cc] * bimg[yi[cc] * kW + xi[cc]];
    rgbb[(p * kNV + v) * 3 + ch] = acc;
  }
  float r1x = X - refpi[3], r1y = Y - refpi[7], r1z = Z - refpi[11];
  float n1 = rsqrtf(r1x * r1x + r1y * r1y + r1z * r1z);
  const float* Pi = spi + v * 16;
  float r2x = X - Pi[3], r2y = Y - Pi[7], r2z = Z - Pi[11];
  float n2 = rsqrtf(r2x * r2x + r2y * r2y + r2z * r2z);
  dirb[(p * kNV + v) * 3 + 0] = r1x * n1 - r2x * n2;
  dirb[(p * kNV + v) * 3 + 1] = r1y * n1 - r2y * n2;
  dirb[(p * kNV + v) * 3 + 2] = r1z * n1 - r2z * n2;
}

// ---------------- 3D trilinear sample (token 0) ------------------------------
__global__ __launch_bounds__(256) void vol_kernel(const float* __restrict__ p3d,
                                                  const float* __restrict__ vol,
                                                  float* __restrict__ X0) {
  int t = blockIdx.x * blockDim.x + threadIdx.x;
  if (t >= kP * kC) return;
  int p = t >> 5, c = t & 31;
  float x = p3d[p * 3], y = p3d[p * 3 + 1], z = p3d[p * 3 + 2];
  float ix = (x + 1.f) * 0.5f * (float)(kR - 1);
  float iy = (y + 1.f) * 0.5f * (float)(kR - 1);
  float iz = (z + 1.f) * 0.5f * (float)(kR - 1);
  float x0 = floorf(ix), y0 = floorf(iy), z0 = floorf(iz);
  float wx = ix - x0, wy = iy - y0, wz = iz - z0;
  const float* vb = vol + (size_t)c * kR * kR * kR;
  float acc = 0.f;
  for (int dz = 0; dz < 2; ++dz)
    for (int dy = 0; dy < 2; ++dy)
      for (int dx = 0; dx < 2; ++dx) {
        int xi = (int)fminf(fmaxf(x0 + (float)dx, 0.f), (float)(kR - 1));
        int yi = (int)fminf(fmaxf(y0 + (float)dy, 0.f), (float)(kR - 1));
        int zi = (int)fminf(fmaxf(z0 + (float)dz, 0.f), (float)(kR - 1));
        float w = (dx ? wx : 1.f - wx) * (dy ? wy : 1.f - wy) * (dz ? wz : 1.f - wz);
        acc += w * vb[(zi * kR + yi) * kR + xi];
      }
  X0[(size_t)p * 288 + c] = acc;
}

// ---------------- dvt: fused linear-attention LoFTR, 1 wave = 1 point --------
// All WMMA outputs land in 16-row padded LDS tiles -> unconditional stores.
__global__ __launch_bounds__(128) void dvt_kernel(
    const float* __restrict__ X0, float* __restrict__ XOut,
    const _Float16* __restrict__ wq, const _Float16* __restrict__ wk,
    const _Float16* __restrict__ wv, const _Float16* __restrict__ wm,
    const _Float16* __restrict__ w1, const _Float16* __restrict__ w2,
    const float* __restrict__ g1, const float* __restrict__ b1,
    const float* __restrict__ g2, const float* __restrict__ b2) {
  __shared__ _Float16 shA[4][16 * 64];
  __shared__ _Float16 shB[4][16 * 64];
  __shared__ float shX[4][9 * 32];
  __shared__ float shQ[4][16 * 32];
  __shared__ float shK[4][16 * 32];
  __shared__ float shV[4][16 * 32];
  __shared__ float shM[4][16 * 32];
  __shared__ float shKV[4][32 * 4];
  __shared__ float shKs[4][32];
  int wid = threadIdx.x >> 5, lane = threadIdx.x & 31;
  int p = blockIdx.x * 4 + wid;
  _Float16* A  = shA[wid]; _Float16* Bh = shB[wid];
  float* Xf = shX[wid]; float* Qf = shQ[wid]; float* Kf = shK[wid];
  float* Vf = shV[wid]; float* Mf = shM[wid];
  float* KVf = shKV[wid]; float* Ksf = shKs[wid];
  const float* xin = X0 + (size_t)p * 288;
  for (int i = lane; i < 512; i += 32) A[i] = (_Float16)0.f;
  for (int i = lane; i < 288; i += 32) { float v = xin[i]; Xf[i] = v; A[i] = (_Float16)v; }
  __syncthreads();
  for (int nt = 0; nt < 2; ++nt) {
    tile_store_full<false,false>(wmma_mk(A, wq, lane, 0, nt, 32), lane, 0, nt, Qf, 32);
    tile_store_full<false,false>(wmma_mk(A, wk, lane, 0, nt, 32), lane, 0, nt, Kf, 32);
    tile_store_full<false,false>(wmma_mk(A, wv, lane, 0, nt, 32), lane, 0, nt, Vf, 32);
  }
  __syncthreads();
  {  // KV and K-sum per (head, d) lane
    int h = lane >> 2, d = lane & 3;
    float kv[4] = {0, 0, 0, 0}, ks = 0;
    for (int s = 0; s < 9; ++s) {
      float kp = elup(Kf[s * 32 + h * 4 + d]);
      ks += kp;
      for (int vv = 0; vv < 4; ++vv) kv[vv] += kp * Vf[s * 32 + h * 4 + vv] * (1.f / 9.f);
    }
    for (int vv = 0; vv < 4; ++vv) KVf[lane * 4 + vv] = kv[vv];
    Ksf[lane] = ks;
  }
  __syncthreads();
  {  // msg per (head, v') lane
    int h = lane >> 2, vv = lane & 3;
    for (int l = 0; l < 9; ++l) {
      float z = 0, m = 0;
      for (int d = 0; d < 4; ++d) {
        float qp = elup(Qf[l * 32 + h * 4 + d]);
        z += qp * Ksf[h * 4 + d];
        m += qp * KVf[(h * 4 + d) * 4 + vv];
      }
      Mf[l * 32 + h * 4 + vv] = m * (1.f / (z + 1e-6f)) * 9.f;
    }
  }
  __syncthreads();
  for (int i = lane; i < 288; i += 32) A[i] = (_Float16)Mf[i];  // rows 9..15 stay 0
  __syncthreads();
  for (int nt = 0; nt < 2; ++nt)
    tile_store_full<false,false>(wmma_mk(A, wm, lane, 0, nt, 32), lane, 0, nt, Qf, 32);
  __syncthreads();
  if (lane < 9) {  // ln1
    float mu = 0; for (int c = 0; c < 32; ++c) mu += Qf[lane * 32 + c];
    mu *= (1.f / 32.f);
    float var = 0;
    for (int c = 0; c < 32; ++c) { float dd = Qf[lane * 32 + c] - mu; var += dd * dd; }
    var *= (1.f / 32.f);
    float rs = rsqrtf(var + 1e-5f);
    for (int c = 0; c < 32; ++c)
      Mf[lane * 32 + c] = (Qf[lane * 32 + c] - mu) * rs * g1[c] + b1[c];
  }
  __syncthreads();
  for (int i = lane; i < 1024; i += 32) A[i] = (_Float16)0.f;  // same-lane indices
  for (int i = lane; i < 288; i += 32) {
    int r = i / 32, c = i - r * 32;        // c == lane pattern -> same-lane overwrite
    A[r * 64 + c]      = (_Float16)Xf[i];
    A[r * 64 + 32 + c] = (_Float16)Mf[i];
  }
  __syncthreads();
  for (int nt = 0; nt < 4; ++nt)
    tile_store_full<true,true>(wmma_mk(A, w1, lane, 0, nt, 64), lane, 0, nt, Bh, 64);
  __syncthreads();
  for (int nt = 0; nt < 2; ++nt)
    tile_store_full<false,false>(wmma_mk(Bh, w2, lane, 0, nt, 64), lane, 0, nt, Qf, 32);
  __syncthreads();
  if (lane < 9) {  // ln2 + residual
    float mu = 0; for (int c = 0; c < 32; ++c) mu += Qf[lane * 32 + c];
    mu *= (1.f / 32.f);
    float var = 0;
    for (int c = 0; c < 32; ++c) { float dd = Qf[lane * 32 + c] - mu; var += dd * dd; }
    var *= (1.f / 32.f);
    float rs = rsqrtf(var + 1e-5f);
    float* xo = XOut + (size_t)p * 288 + lane * 32;
    for (int c = 0; c < 32; ++c)
      xo[c] = Xf[lane * 32 + c] + (Qf[lane * 32 + c] - mu) * rs * g2[c] + b2[c];
  }
}

// ---------------- view-weight MLP + blend + fuse + posenc + drt token assembly
__global__ __launch_bounds__(256) void fuse_kernel(
    const float* __restrict__ XO, const float* __restrict__ X0,
    const float* __restrict__ rgbb, const float* __restrict__ maskb,
    const float* __restrict__ dirb, const float* __restrict__ zv,
    const float* __restrict__ w1, const float* __restrict__ b1,
    const float* __restrict__ w2, const float* __restrict__ b2,
    const float* __restrict__ w3, const float* __restrict__ b3,
    const float* __restrict__ fw, const float* __restrict__ fb,
    const float* __restrict__ vtok, float* __restrict__ Xd) {
  int p = blockIdx.x * blockDim.x + threadIdx.x;
  if (p >= kP) return;
  float xw[kNV];
  for (int v = 0; v < kNV; ++v) {
    float in[35];
    const float* vf = XO + ((size_t)p * 9 + 1 + v) * 32;
    for (int c = 0; c < 32; ++c) in[c] = vf[c];
    for (int i = 0; i < 3; ++i) in[32 + i] = dirb[(p * kNV + v) * 3 + i];
    float h1[16];
    for (int o = 0; o < 16; ++o) {
      float a = b1[o];
      for (int i = 0; i < 35; ++i) a += w1[o * 35 + i] * in[i];
      h1[o] = fmaxf(a, 0.f);
    }
    float h2[8];
    for (int o = 0; o < 8; ++o) {
      float a = b2[o];
      for (int i = 0; i < 16; ++i) a += w2[o * 16 + i] * h1[i];
      h2[o] = fmaxf(a, 0.f);
    }
    float a = b3[0];
    for (int i = 0; i < 8; ++i) a += w3[i] * h2[i];
    xw[v] = (maskb[p * kNV + v] == 0.f) ? -1e9f : a;
  }
  float mx = xw[0];
  for (int v = 1; v < kNV; ++v) mx = fmaxf(mx, xw[v]);
  float sum = 0.f;
  for (int v = 0; v < kNV; ++v) { xw[v] = expf(xw[v] - mx); sum += xw[v]; }
  float inv = 1.f / sum;
  float rad[35];
  for (int i = 0; i < 35; ++i) rad[i] = 0.f;
  for (int v = 0; v < kNV; ++v) {
    float wv = xw[v] * inv;
    for (int c = 0; c < 3; ++c) rad[c] += wv * rgbb[(p * kNV + v) * 3 + c];
    const float* of = X0 + ((size_t)p * 9 + 1 + v) * 32;  // ORIGINAL sampled feats
    for (int c = 0; c < 32; ++c) rad[3 + c] += wv * of[c];
  }
  int ray = p / kSN, s = p - (p / kSN) * kSN;
  float* xo = Xd + ((size_t)ray * kL + 1 + s) * kD;
  const float* x0t = XO + (size_t)p * 9 * 32;  // updated token 0
  for (int c = 0; c < 32; ++c) xo[c] = x0t[c];
  for (int o = 0; o < 32; ++o) {
    float a = fb[o];
    for (int i = 0; i < 35; ++i) a += fw[o * 35 + i] * rad[i];
    xo[32 + o] = a;
  }
  float z = 100.f * zv[p];
  const float divt[4] = {1.f, 0.1f, 0.01f, 0.001f};
  for (int j = 0; j < 4; ++j) {
    xo[64 + 2 * j]     = sinf(z * divt[j]);
    xo[64 + 2 * j + 1] = cosf(z * divt[j]);
  }
  if (s == 0) {
    float* t0 = Xd + (size_t)ray * kL * kD;
    for (int d2 = 0; d2 < kD; ++d2) t0[d2] = vtok[d2];
  }
}

// ---------------- drt: fused causal full-attention LoFTR, 1 block = 1 ray ----
// Dynamic LDS arena (~165 KB, fits 320 KB WGP LDS). All WMMA stores land in
// tile-padded buffers (ld 80 / 160) -> fully unconditional epilogues.
__global__ __launch_bounds__(128) void drt_kernel(
    float* __restrict__ Xd,
    const _Float16* __restrict__ wq, const _Float16* __restrict__ wk,
    const _Float16* __restrict__ wv, const _Float16* __restrict__ wm,
    const _Float16* __restrict__ w1, const _Float16* __restrict__ w2,
    const float* __restrict__ g1, const float* __restrict__ b1,
    const float* __restrict__ g2, const float* __restrict__ b2) {
  extern __shared__ char smem[];
  float*    sX  = (float*)(smem);                 // 65x72 f32   (18720)
  _Float16* sXh = (_Float16*)(smem + 18720);      // 80x96 f16   (15360)
  _Float16* sQh = (_Float16*)(smem + 34080);      // 80x80 f16   (12800)
  _Float16* sKh = (_Float16*)(smem + 46880);      // 80x80 f16
  _Float16* sVh = (_Float16*)(smem + 59680);      // 80x80 f16
  _Float16* sMh = (_Float16*)(smem + 72480);      // 80x96 f16   (15360)
  float*    sM2 = (float*)(smem + 87840);         // 80x80 f32   (25600)
  _Float16* sCt = (_Float16*)(smem + 113440);     // 80x160 f16  (25600)
  _Float16* sH1 = (_Float16*)(smem + 139040);     // 80x160 f16  (25600) end 164640
  int tid = threadIdx.x, wid = tid >> 5, lane = tid & 31;
  size_t ray = blockIdx.x;
  float* xg = Xd + ray * (kL * kD);
  for (int i = tid; i < 80 * 96; i += 128) { sXh[i] = (_Float16)0.f; sMh[i] = (_Float16)0.f; }
  for (int i = tid; i < kL * kD; i += 128) sX[i] = xg[i];
  __syncthreads();
  for (int i = tid; i < kL * kD; i += 128) {
    int r = i / kD, c = i - r * kD;
    sXh[r * 96 + c] = (_Float16)sX[i];
  }
  __syncthreads();
  for (int rt = wid; rt < 5; rt += 4)
    for (int nt = 0; nt < 5; ++nt) {
      tile_store_full<false,true>(wmma_mk(sXh, wq, lane, rt, nt, 96), lane, rt, nt, sQh, 80);
      tile_store_full<false,true>(wmma_mk(sXh, wk, lane, rt, nt, 96), lane, rt, nt, sKh, 80);
      tile_store_full<false,true>(wmma_mk(sXh, wv, lane, rt, nt, 96), lane, rt, nt, sVh, 80);
    }
  __syncthreads();
  for (int idx = tid; idx < kL * 8; idx += 128) {  // causal softmax attention
    int l = idx >> 3, h = idx & 7, base = h * 9;
    float mx = -1e30f;
    for (int s = 0; s <= l; ++s) {
      float sc = 0;
      for (int j = 0; j < 9; ++j)
        sc += (float)sQh[l * 80 + base + j] * (float)sKh[s * 80 + base + j];
      mx = fmaxf(mx, sc * (1.f / 3.f));
    }
    float sum = 0; float acc[9] = {0, 0, 0, 0, 0, 0, 0, 0, 0};
    for (int s = 0; s <= l; ++s) {
      float sc = 0;
      for (int j = 0; j < 9; ++j)
        sc += (float)sQh[l * 80 + base + j] * (float)sKh[s * 80 + base + j];
      float e = expf(sc * (1.f / 3.f) - mx);
      sum += e;
      for (int j = 0; j < 9; ++j) acc[j] += e * (float)sVh[s * 80 + base + j];
    }
    float inv = 1.f / sum;
    for (int j = 0; j < 9; ++j) sMh[l * 96 + base + j] = (_Float16)(acc[j] * inv);
  }
  __syncthreads();
  for (int rt = wid; rt < 5; rt += 4)
    for (int nt = 0; nt < 5; ++nt)
      tile_store_full<false,false>(wmma_mk(sMh, wm, lane, rt, nt, 96), lane, rt, nt, sM2, 80);
  __syncthreads();
  for (int l = tid; l < kL; l += 128) {  // ln1 (valid 65x72 region, ld 80)
    float mu = 0; for (int c = 0; c < kD; ++c) mu += sM2[l * 80 + c];
    mu *= (1.f / (float)kD);
    float var = 0;
    for (int c = 0; c < kD; ++c) { float dd = sM2[l * 80 + c] - mu; var += dd * dd; }
    var *= (1.f / (float)kD);
    float rs = rsqrtf(var + 1e-5f);
    for (int c = 0; c < kD; ++c) sM2[l * 80 + c] = (sM2[l * 80 + c] - mu) * rs * g1[c] + b1[c];
  }
  __syncthreads();
  for (int i = tid; i < 80 * 160; i += 128) { sCt[i] = (_Float16)0.f; sH1[i] = (_Float16)0.f; }
  __syncthreads();
  for (int i = tid; i < kL * kD; i += 128) {
    int r = i / kD, c = i - r * kD;
    sCt[r * 160 + c]      = (_Float16)sX[i];
    sCt[r * 160 + 72 + c] = (_Float16)sM2[r * 80 + c];
  }
  __syncthreads();
  for (int rt = wid; rt < 5; rt += 4)
    for (int nt = 0; nt < 9; ++nt)
      tile_store_full<true,true>(wmma_mk(sCt, w1, lane, rt, nt, 160), lane, rt, nt, sH1, 160);
  __syncthreads();
  for (int rt = wid; rt < 5; rt += 4)
    for (int nt = 0; nt < 5; ++nt)
      tile_store_full<false,false>(wmma_mk(sH1, w2, lane, rt, nt, 160), lane, rt, nt, sM2, 80);
  __syncthreads();
  for (int l = tid; l < kL; l += 128) {  // ln2 + residual, in-place writeback
    float mu = 0; for (int c = 0; c < kD; ++c) mu += sM2[l * 80 + c];
    mu *= (1.f / (float)kD);
    float var = 0;
    for (int c = 0; c < kD; ++c) { float dd = sM2[l * 80 + c] - mu; var += dd * dd; }
    var *= (1.f / (float)kD);
    float rs = rsqrtf(var + 1e-5f);
    for (int c = 0; c < kD; ++c)
      xg[l * kD + c] = sX[l * kD + c] + (sM2[l * 80 + c] - mu) * rs * g2[c] + b2[c];
  }
}

// ---------------- decoder layer (1 query) + radiance MLP, 1 block = 1 ray ----
__global__ __launch_bounds__(128) void dec_kernel(
    const float* __restrict__ Xd,
    const float* __restrict__ Wq, const float* __restrict__ Wk,
    const float* __restrict__ Wv, const float* __restrict__ Wm,
    const float* __restrict__ W1, const float* __restrict__ W2,
    const float* __restrict__ g1, const float* __restrict__ b1,
    const float* __restrict__ g2, const float* __restrict__ b2,
    const float* __restrict__ rw1, const float* __restrict__ rb1,
    const float* __restrict__ rw2, const float* __restrict__ rb2,
    const float* __restrict__ rw3, const float* __restrict__ rb3,
    float* __restrict__ outRad, float* __restrict__ outWt) {
  extern __shared__ char smem[];
  float* sSrc = (float*)smem;                  // 64x72
  float* sK   = (float*)(smem + 18432);
  float* sV   = (float*)(smem + 36864);
  float* sX0  = (float*)(smem + 55296);        // 72
  float* sQv  = (float*)(smem + 55584);        // 72 (also reused as out token)
  float* sMsg = (float*)(smem + 55872);        // 72
  float* sMn  = (float*)(smem + 56160);        // 72
  float* sCat = (float*)(smem + 56448);        // 144
  float* sHH  = (float*)(smem + 57024);        // 144
  float* sM2  = (float*)(smem + 57600);        // 72
  float* sSc  = (float*)(smem + 57888);        // 64
  float* sR1  = (float*)(smem + 58144);        // 32
  float* sR2  = (float*)(smem + 58272);        // 16
  float* sSt  = (float*)(smem + 58336);        // 2
  int tid = threadIdx.x;
  size_t ray = blockIdx.x;
  const float* xg = Xd + ray * (kL * kD);
  for (int i = tid; i < kD; i += 128) sX0[i] = xg[i];
  for (int i = tid; i < 64 * kD; i += 128) sSrc[i] = xg[kD + i];
  __syncthreads();
  for (int o = tid; o < kD; o += 128) {
    float a = 0;
    for (int i = 0; i < kD; ++i) a += sX0[i] * Wq[o * kD + i];
    sQv[o] = a;
  }
  for (int idx = tid; idx < 64 * kD; idx += 128) {
    int s = idx / kD, o = idx - s * kD;
    float ak = 0, av = 0;
    for (int i = 0; i < kD; ++i) {
      float xv = sSrc[s * kD + i];
      ak += xv * Wk[o * kD + i];
      av += xv * Wv[o * kD + i];
    }
    sK[idx] = ak; sV[idx] = av;
  }
  __syncthreads();
  if (tid < 64) {
    float a = 0;
    for (int i = 0; i < kD; ++i) a += sQv[i] * sK[tid * kD + i];
    sSc[tid] = a * rsqrtf(72.f);
  }
  __syncthreads();
  if (tid == 0) {
    float mx = -1e30f;
    for (int s = 0; s < 64; ++s) mx = fmaxf(mx, sSc[s]);
    float sum = 0;
    for (int s = 0; s < 64; ++s) { float e = expf(sSc[s] - mx); sSc[s] = e; sum += e; }
    float inv = 1.f / sum;
    for (int s = 0; s < 64; ++s) { sSc[s] *= inv; outWt[ray * 64 + s] = sSc[s]; }
  }
  __syncthreads();
  for (int o = tid; o < kD; o += 128) {
    float a = 0;
    for (int s = 0; s < 64; ++s) a += sSc[s] * sV[s * kD + o];
    sMsg[o] = a;
  }
  __syncthreads();
  for (int o = tid; o < kD; o += 128) {
    float a = 0;
    for (int i = 0; i < kD; ++i) a += sMsg[i] * Wm[o * kD + i];
    sM2[o] = a;
  }
  __syncthreads();
  if (tid == 0) {
    float mu = 0; for (int c = 0; c < kD; ++c) mu += sM2[c];
    mu *= (1.f / (float)kD);
    float var = 0;
    for (int c = 0; c < kD; ++c) { float dd = sM2[c] - mu; var += dd * dd; }
    sSt[0] = mu; sSt[1] = var * (1.f / (float)kD);
  }
  __syncthreads();
  for (int o = tid; o < kD; o += 128)
    sMn[o] = (sM2[o] - sSt[0]) * rsqrtf(sSt[1] + 1e-5f) * g1[o] + b1[o];
  __syncthreads();
  for (int o = tid; o < kD; o += 128) { sCat[o] = sX0[o]; sCat[72 + o] = sMn[o]; }
  __syncthreads();
  for (int o = tid; o < 144; o += 128) {
    float a = 0;
    for (int i = 0; i < 144; ++i) a += sCat[i] * W1[o * 144 + i];
    sHH[o] = fmaxf(a, 0.f);
  }
  __syncthreads();
  for (int o = tid; o < kD; o += 128) {
    float a = 0;
    for (int i = 0; i < 144; ++i) a += sHH[i] * W2[o * 144 + i];
    sM2[o] = a;
  }
  __syncthreads();
  if (tid == 0) {
    float mu = 0; for (int c = 0; c < kD; ++c) mu += sM2[c];
    mu *= (1.f / (float)kD);
    float var = 0;
    for (int c = 0; c < kD; ++c) { float dd = sM2[c] - mu; var += dd * dd; }
    sSt[0] = mu; sSt[1] = var * (1.f / (float)kD);
  }
  __syncthreads();
  for (int o = tid; o < kD; o += 128)
    sQv[o] = sX0[o] + (sM2[o] - sSt[0]) * rsqrtf(sSt[1] + 1e-5f) * g2[o] + b2[o];
  __syncthreads();
  for (int o = tid; o < 32; o += 128) {
    float a = rb1[o];
    for (int i = 0; i < kD; ++i) a += rw1[o * kD + i] * sQv[i];
    sR1[o] = fmaxf(a, 0.f);
  }
  __syncthreads();
  for (int o = tid; o < 16; o += 128) {
    float a = rb2[o];
    for (int i = 0; i < 32; ++i) a += rw2[o * 32 + i] * sR1[i];
    sR2[o] = fmaxf(a, 0.f);
  }
  __syncthreads();
  for (int o = tid; o < 3; o += 128) {
    float a = rb3[o];
    for (int i = 0; i < 16; ++i) a += rw3[o * 16 + i] * sR2[i];
    outRad[ray * 3 + o] = 1.f / (1.f + expf(-a));
  }
}

// ---------------- host orchestration ----------------------------------------
extern "C" void kernel_launch(void* const* d_in, const int* in_sizes, int n_in,
                              void* d_out, int out_size, void* d_ws, size_t ws_size,
                              hipStream_t stream) {
  (void)in_sizes; (void)n_in; (void)out_size; (void)ws_size;
  const float* p3d   = (const float*)d_in[0];
  const float* simg  = (const float*)d_in[1];
  const float* sfeat = (const float*)d_in[2];
  const float* refpi = (const float*)d_in[3];
  const float* spi   = (const float*)d_in[4];
  const float* sp    = (const float*)d_in[5];
  const float* zv    = (const float*)d_in[6];
  const float* vol   = (const float*)d_in[7];
  // params pytree leaves (sorted nested keys): dec(0-9) drt(10-19) dvt(20-29)
  // fuse_b(30) fuse_w(31) rad(b1,b2,b3,w1,w2,w3 = 32..37) rw(38..43) view_token(44)
  auto PAR = [&](int i) { return (const float*)d_in[8 + i]; };

  char* ws = (char*)d_ws;
  constexpr size_t SZ_X0   = (size_t)kP * 9 * 32 * 4;   // 151 MB
  constexpr size_t OFF_X0  = 0;
  constexpr size_t OFF_XO  = OFF_X0 + SZ_X0;
  constexpr size_t OFF_RGB = OFF_XO + SZ_X0;
  constexpr size_t OFF_MSK = OFF_RGB + (size_t)kP * kNV * 3 * 4;
  constexpr size_t OFF_DIR = OFF_MSK + (size_t)kP * kNV * 4;
  constexpr size_t OFF_XDR = OFF_DIR + (size_t)kP * kNV * 3 * 4;
  size_t off = OFF_XDR + (size_t)kRN * kL * kD * 4;

  auto pk = [&](int pi, int N, int K, int Np, int Kp) -> _Float16* {
    _Float16* dst = (_Float16*)(ws + off);
    int tot = Np * Kp;
    pack_w_kernel<<<(tot + 255) / 256, 256, 0, stream>>>(PAR(pi), dst, N, K, Np, Kp);
    off += (size_t)tot * 2;
    off = (off + 255) & ~(size_t)255;
    return dst;
  };
  _Float16* dvtQ  = pk(28, 32, 32, 32, 32);
  _Float16* dvtK  = pk(20, 32, 32, 32, 32);
  _Float16* dvtV  = pk(29, 32, 32, 32, 32);
  _Float16* dvtM  = pk(25, 32, 32, 32, 32);
  _Float16* dvtW1 = pk(26, 64, 64, 64, 64);
  _Float16* dvtW2 = pk(27, 32, 64, 32, 64);
  _Float16* drtQ  = pk(18, 72, 72, 80, 96);
  _Float16* drtK  = pk(10, 72, 72, 80, 96);
  _Float16* drtV  = pk(19, 72, 72, 80, 96);
  _Float16* drtM  = pk(15, 72, 72, 80, 96);
  _Float16* drtW1 = pk(16, 144, 144, 144, 160);
  _Float16* drtW2 = pk(17, 72, 144, 80, 160);

  float* X0    = (float*)(ws + OFF_X0);
  float* XO    = (float*)(ws + OFF_XO);
  float* RGBB  = (float*)(ws + OFF_RGB);
  float* MASKB = (float*)(ws + OFF_MSK);
  float* DIRB  = (float*)(ws + OFF_DIR);
  float* XDRT  = (float*)(ws + OFF_XDR);

  float* outRad = (float*)d_out;                    // (2048,1,3)
  float* outWt  = (float*)d_out + 6144;             // (2048,64)
  float* outPp  = (float*)d_out + 6144 + kP;        // (1,8,2,2048,64)

  geom_kernel<<<(kP * kNV) / 256, 256, 0, stream>>>(p3d, simg, sfeat, refpi, spi,
                                                    sp, X0, RGBB, MASKB, DIRB, outPp);
  vol_kernel<<<(kP * kC) / 256, 256, 0, stream>>>(p3d, vol, X0);
  dvt_kernel<<<kP / 4, 128, 0, stream>>>(X0, XO, dvtQ, dvtK, dvtV, dvtM, dvtW1,
                                         dvtW2, PAR(22), PAR(21), PAR(24), PAR(23));
  fuse_kernel<<<kP / 256, 256, 0, stream>>>(XO, X0, RGBB, MASKB, DIRB, zv,
                                            PAR(41), PAR(38), PAR(42), PAR(39),
                                            PAR(43), PAR(40), PAR(31), PAR(30),
                                            PAR(44), XDRT);
  drt_kernel<<<kRN, 128, 164864, stream>>>(XDRT, drtQ, drtK, drtV, drtM, drtW1,
                                           drtW2, PAR(12), PAR(11), PAR(14), PAR(13));
  dec_kernel<<<kRN, 128, 58368, stream>>>(XDRT, PAR(8), PAR(0), PAR(9), PAR(5),
                                          PAR(6), PAR(7), PAR(2), PAR(1), PAR(4),
                                          PAR(3), PAR(35), PAR(32), PAR(36),
                                          PAR(33), PAR(37), PAR(34), outRad, outWt);
}
